// AdMSoftmaxLoss_4853313045243
// MI455X (gfx1250) — matrix operand order repaired and checked by reading, not compiled, not run on hardware
//
#include <hip/hip_runtime.h>
#include <hip/hip_bf16.h>
#include <math.h>

// ---------------------------------------------------------------------------
// AdMSoftmax loss, fused for MI455X (gfx1250, wave32, WMMA).
//   N=8192 samples, D=1024 features, C=10240 classes, S=30, ml=0.4, ms=0.1
// 172-GFLOP GEMM as bf16x3 split-precision WMMA (16x16x32 bf16), fused
// clamp+exp+row-sum epilogue (320MB logits never hit memory), global->LDS
// staged with gfx1250 async-to-LDS copies (ASYNCcnt) and double buffering.
// ---------------------------------------------------------------------------

#define N_ROWS 8192
#define D_DIM  1024
#define C_DIM  10240
#define S_SCALE 30.0f

typedef __bf16 v16bf __attribute__((ext_vector_type(16)));
typedef float  v8f   __attribute__((ext_vector_type(8)));

struct Frag32B { uint4 q0, q1; };   // 32 bytes == one v16bf fragment

__device__ __forceinline__ unsigned short f32_to_bf16_rne(float f) {
  unsigned int u = __float_as_uint(f);
  unsigned int r = u + 0x7FFFu + ((u >> 16) & 1u);
  return (unsigned short)(r >> 16);
}
__device__ __forceinline__ float bf16_to_f32(unsigned short h) {
  return __uint_as_float(((unsigned int)h) << 16);
}

// gfx1250 async DMA: 16B global -> LDS, GVS mode (SGPR64 base + per-lane i32
// byte offset), tracked on ASYNCcnt.  ldsAddr = low 32 bits of flat LDS ptr
// (== wave-relative LDS byte offset per the aperture mapping).
__device__ __forceinline__ void async_b128(unsigned ldsAddr,
                                           const unsigned short* sbase,
                                           int gByteOff) {
  asm volatile("global_load_async_to_lds_b128 %0, %1, %2"
               :: "v"(ldsAddr), "v"(gByteOff), "s"(sbase)
               : "memory");
}

// --- 1) L2-normalize rows (eps=1e-12, matches F.normalize) and split to
//        bf16 hi/lo pair; also store 1/||row|| for the exact-target pass.
__global__ __launch_bounds__(256) void norm_split_kernel(
    const float* __restrict__ src, unsigned short* __restrict__ hi,
    unsigned short* __restrict__ lo, float* __restrict__ inv) {
  __shared__ float red[256];
  __shared__ float s_inv;
  const int row = blockIdx.x;
  const float* v = src + (size_t)row * D_DIM;
  float vals[4];
  float ss = 0.f;
#pragma unroll
  for (int i = 0; i < 4; ++i) {
    vals[i] = v[threadIdx.x + i * 256];
    ss += vals[i] * vals[i];
  }
  red[threadIdx.x] = ss;
  __syncthreads();
  for (int s = 128; s > 0; s >>= 1) {
    if (threadIdx.x < s) red[threadIdx.x] += red[threadIdx.x + s];
    __syncthreads();
  }
  if (threadIdx.x == 0) {
    s_inv = 1.0f / fmaxf(sqrtf(red[0]), 1e-12f);
    inv[row] = s_inv;
  }
  __syncthreads();
  const float iv = s_inv;
#pragma unroll
  for (int i = 0; i < 4; ++i) {
    float xn = vals[i] * iv;
    unsigned short h = f32_to_bf16_rne(xn);
    float rem = xn - bf16_to_f32(h);
    size_t idx = (size_t)row * D_DIM + threadIdx.x + i * 256;
    hi[idx] = h;
    lo[idx] = f32_to_bf16_rne(rem);
  }
}

__global__ void zero_f32_kernel(float* __restrict__ p, int n) {
  int i = blockIdx.x * 256 + threadIdx.x;
  if (i < n) p[i] = 0.f;
}

// --- 2) Fused bf16x3 WMMA GEMM + clamp + exp + per-row sum.
//        Block tile 128(N) x 128(C), K-step 32, 8 waves (2x4) of 64x32 each.
//        Double-buffered LDS filled by async-to-LDS DMA.
#define BK  32
#define LDT 40                      // padded LDS row stride (elems): 80B
#define TILE_ELEMS   (128 * LDT)    // 5120 elems = 10240 B per array
#define TILE_BYTES   (TILE_ELEMS * 2)
#define STAGE_ELEMS  (4 * TILE_ELEMS)
#define STAGE_BYTES  (STAGE_ELEMS * 2)         // 40960 B
#define SMEM_BYTES   (2 * STAGE_BYTES)         // 81920 B dynamic LDS

__global__ __launch_bounds__(256) void gemm_expsum_kernel(
    const unsigned short* __restrict__ xh, const unsigned short* __restrict__ xl,
    const unsigned short* __restrict__ wh, const unsigned short* __restrict__ wl,
    float* __restrict__ rowsum) {
  extern __shared__ unsigned short smem[];   // [2][4][128*LDT]

  const int tid  = threadIdx.x;
  const int lane = tid & 31;
  const int wave = tid >> 5;
  const int wrow = wave & 1;    // 2 row-slabs of 64
  const int wcol = wave >> 1;   // 4 col-slabs of 32
  const int half = lane >> 4;   // wave32 half-group
  const int l15  = lane & 15;

  const int rowBase = blockIdx.y * 128;   // N
  const int colBase = blockIdx.x * 128;   // C

  const unsigned ldsBase = (unsigned)(uintptr_t)smem;

  v8f acc[4][2];
#pragma unroll
  for (int mt = 0; mt < 4; ++mt)
#pragma unroll
    for (int nt = 0; nt < 2; ++nt)
      acc[mt][nt] = (v8f){0.f, 0.f, 0.f, 0.f, 0.f, 0.f, 0.f, 0.f};

  // Issue one K-stage (8 async-to-LDS b128 per wave: 4 arrays x 2 chunks).
  auto issueStage = [&](int k0, int buf) {
#pragma unroll
    for (int it = 0; it < 2; ++it) {
      int c = tid + it * 256;           // 512 chunks of 8 elems (16B)
      int r = c >> 2, cc = c & 3;
      unsigned lA = ldsBase + (unsigned)(buf * STAGE_BYTES + (r * LDT + cc * 8) * 2);
      int gA = ((rowBase + r) * D_DIM + k0 + cc * 8) * 2;   // bytes
      int gB = ((colBase + r) * D_DIM + k0 + cc * 8) * 2;   // bytes
      async_b128(lA,                  xh, gA);
      async_b128(lA + TILE_BYTES,     xl, gA);
      async_b128(lA + 2 * TILE_BYTES, wh, gB);
      async_b128(lA + 3 * TILE_BYTES, wl, gB);
    }
  };

  issueStage(0, 0);

  for (int ks = 0; ks < D_DIM / BK; ++ks) {
    if (ks + 1 < D_DIM / BK) {
      issueStage((ks + 1) * BK, (ks + 1) & 1);        // prefetch next stage
      asm volatile("s_wait_asynccnt 0x8" ::: "memory"); // stage ks landed
    } else {
      asm volatile("s_wait_asynccnt 0x0" ::: "memory");
    }
    __syncthreads();   // stage ks visible to all waves

    const unsigned short* sAh = smem + (ks & 1) * STAGE_ELEMS;
    const unsigned short* sAl = sAh + TILE_ELEMS;
    const unsigned short* sBh = sAh + 2 * TILE_ELEMS;
    const unsigned short* sBl = sAh + 3 * TILE_ELEMS;

    // B fragments (32x16): K = {base..base+7, base+16..base+23}, base=half*8, col l15.
    v16bf bh[2], bl[2];
#pragma unroll
    for (int nt = 0; nt < 2; ++nt) {
      int bRow = wcol * 32 + nt * 16 + l15;
      int off = bRow * LDT + half * 8;
      Frag32B qh, ql;
      qh.q0 = *(const uint4*)(&sBh[off]);
      qh.q1 = *(const uint4*)(&sBh[off + 16]);
      ql.q0 = *(const uint4*)(&sBl[off]);
      ql.q1 = *(const uint4*)(&sBl[off + 16]);
      bh[nt] = __builtin_bit_cast(v16bf, qh);
      bl[nt] = __builtin_bit_cast(v16bf, ql);
    }
#pragma unroll
    for (int mt = 0; mt < 4; ++mt) {
      // A fragments (16x32): K = {base..base+7, base+16..base+23}, base=half*8.
      int aRow = wrow * 64 + mt * 16 + l15;
      int off = aRow * LDT + half * 8;
      Frag32B qh, ql;
      qh.q0 = *(const uint4*)(&sAh[off]);
      qh.q1 = *(const uint4*)(&sAh[off + 16]);
      ql.q0 = *(const uint4*)(&sAl[off]);
      ql.q1 = *(const uint4*)(&sAl[off + 16]);
      v16bf ah = __builtin_bit_cast(v16bf, qh);
      v16bf al = __builtin_bit_cast(v16bf, ql);
#pragma unroll
      for (int nt = 0; nt < 2; ++nt) {
        // bf16x3 split-precision: hi*hi + lo*hi + hi*lo, f32 accumulate
        acc[mt][nt] = __builtin_amdgcn_wmma_f32_16x16x32_bf16(
            false, ah, false, bh[nt], (short)0, acc[mt][nt], false, false);
        acc[mt][nt] = __builtin_amdgcn_wmma_f32_16x16x32_bf16(
            false, al, false, bh[nt], (short)0, acc[mt][nt], false, false);
        acc[mt][nt] = __builtin_amdgcn_wmma_f32_16x16x32_bf16(
            false, ah, false, bl[nt], (short)0, acc[mt][nt], false, false);
      }
    }
    __syncthreads();   // all waves done with stage ks before its buffer refills
  }

  // Epilogue: clamp, exp(S*logit) (no max-shift needed: logit<=1 => sum<=1e17),
  // reduce over this wave's 32 columns, one atomic per row per wave.
#pragma unroll
  for (int mt = 0; mt < 4; ++mt) {
#pragma unroll
    for (int r = 0; r < 8; ++r) {
      float e = 0.f;
#pragma unroll
      for (int nt = 0; nt < 2; ++nt) {
        float v = acc[mt][nt][r];
        v = fminf(1.0f, fmaxf(-1.0f, v));
        e += __expf(S_SCALE * v);
      }
      // xor masks 1/2/4/8 stay inside each 16-lane half (C/D layout rows differ
      // between halves: M = r + 8*half)
      e += __shfl_xor(e, 1, 32);
      e += __shfl_xor(e, 2, 32);
      e += __shfl_xor(e, 4, 32);
      e += __shfl_xor(e, 8, 32);
      if (l15 == 0) {
        int row = rowBase + wrow * 64 + mt * 16 + r + half * 8;
        atomicAdd(&rowsum[row], e);
      }
    }
  }
}

// --- 3) Exact fp32 target logit: wf[i, labels[i]] via full-precision dot.
__global__ __launch_bounds__(256) void target_kernel(
    const float* __restrict__ x, const float* __restrict__ w,
    const int* __restrict__ labels, const float* __restrict__ xinv,
    const float* __restrict__ winv, float* __restrict__ target) {
  __shared__ float red[256];
  const int row = blockIdx.x;
  const int lab = labels[row];
  const float* xr = x + (size_t)row * D_DIM;
  const float* wr = w + (size_t)lab * D_DIM;
  float d = 0.f;
  for (int i = threadIdx.x; i < D_DIM; i += 256) d += xr[i] * wr[i];
  red[threadIdx.x] = d;
  __syncthreads();
  for (int s = 128; s > 0; s >>= 1) {
    if (threadIdx.x < s) red[threadIdx.x] += red[threadIdx.x + s];
    __syncthreads();
  }
  if (threadIdx.x == 0) {
    float t = red[0] * xinv[row] * winv[lab];
    target[row] = fminf(1.0f, fmaxf(-1.0f, t));
  }
}

// --- 4) Final loss: L_i = numer - log(exp(numer) + (rowsum - exp(S*t))),
//        out = -mean(L).
__global__ __launch_bounds__(256) void loss_kernel(
    const int* __restrict__ labels, const float* __restrict__ target,
    const float* __restrict__ rowsum, float* __restrict__ out) {
  __shared__ float red[256];
  float acc = 0.f;
  for (int i = threadIdx.x; i < N_ROWS; i += 256) {
    float t = target[i];
    float m = (labels[i] <= 5) ? 0.4f : 0.1f;
    float numer = S_SCALE * (t - m);
    float excl = rowsum[i] - expf(S_SCALE * t);
    float denom = expf(numer) + excl;
    acc += numer - logf(denom);
  }
  red[threadIdx.x] = acc;
  __syncthreads();
  for (int s = 128; s > 0; s >>= 1) {
    if (threadIdx.x < s) red[threadIdx.x] += red[threadIdx.x + s];
    __syncthreads();
  }
  if (threadIdx.x == 0) out[0] = -red[0] / (float)N_ROWS;
}

extern "C" void kernel_launch(void* const* d_in, const int* in_sizes, int n_in,
                              void* d_out, int out_size, void* d_ws, size_t ws_size,
                              hipStream_t stream) {
  const float* x      = (const float*)d_in[0];   // (N, D)
  const int*   labels = (const int*)d_in[1];     // (N,)
  const float* w      = (const float*)d_in[2];   // (C, D)

  // Workspace layout (~75.6 MB)
  char* ws = (char*)d_ws;
  size_t oXH = 0;
  size_t oXL = oXH + (size_t)N_ROWS * D_DIM * 2;
  size_t oWH = oXL + (size_t)N_ROWS * D_DIM * 2;
  size_t oWL = oWH + (size_t)C_DIM * D_DIM * 2;
  size_t oXI = oWL + (size_t)C_DIM * D_DIM * 2;
  size_t oWI = oXI + (size_t)N_ROWS * 4;
  size_t oRS = oWI + (size_t)C_DIM * 4;
  size_t oTG = oRS + (size_t)N_ROWS * 4;

  unsigned short* xh = (unsigned short*)(ws + oXH);
  unsigned short* xl = (unsigned short*)(ws + oXL);
  unsigned short* wh = (unsigned short*)(ws + oWH);
  unsigned short* wl = (unsigned short*)(ws + oWL);
  float* xinv   = (float*)(ws + oXI);
  float* winv   = (float*)(ws + oWI);
  float* rowsum = (float*)(ws + oRS);
  float* target = (float*)(ws + oTG);

  norm_split_kernel<<<N_ROWS, 256, 0, stream>>>(x, xh, xl, xinv);
  norm_split_kernel<<<C_DIM, 256, 0, stream>>>(w, wh, wl, winv);
  zero_f32_kernel<<<(N_ROWS + 255) / 256, 256, 0, stream>>>(rowsum, N_ROWS);
  gemm_expsum_kernel<<<dim3(C_DIM / 128, N_ROWS / 128), 256, SMEM_BYTES, stream>>>(
      xh, xl, wh, wl, rowsum);
  target_kernel<<<N_ROWS, 256, 0, stream>>>(x, w, labels, xinv, winv, target);
  loss_kernel<<<1, 256, 0, stream>>>(labels, target, rowsum, (float*)d_out);
}
